// NL1D1D_78529182040831
// MI455X (gfx1250) — matrix operand order repaired and checked by reading, not compiled, not run on hardware
//
#include <hip/hip_runtime.h>

// Problem constants (from reference)
#define B_   8
#define C_   64
#define H_   4096
#define MID_ 32

typedef _Float16 v8h  __attribute__((ext_vector_type(8)));
typedef _Float16 v16h __attribute__((ext_vector_type(16)));
typedef float    v8f  __attribute__((ext_vector_type(8)));

// -----------------------------------------------------------------------------
// Kernel 1: L2-normalize over channels, emit xnh [B,H,64] f16 (row-major per
// (b,h) so it serves as WMMA A and B operand of Q·K^T with contiguous loads),
// and the value projection vth [B,32,H] f16 (V transposed so PV's B operand is
// a contiguous 32B-per-lane load).
// -----------------------------------------------------------------------------
__global__ __launch_bounds__(256) void prep_kernel(
    const float* __restrict__ x,      // [B,C,H]
    const float* __restrict__ g_w,    // [MID,C]
    const float* __restrict__ g_b,    // [MID]
    _Float16* __restrict__ xnh,       // [B,H,C]
    _Float16* __restrict__ vth)       // [B,MID,H]
{
    __shared__ float gw[MID_ * C_];   // 8 KB
    for (int i = threadIdx.x; i < MID_ * C_; i += 256) gw[i] = g_w[i];
    __syncthreads();

    const int g = blockIdx.x * 256 + threadIdx.x;
    const int b = g / H_;
    const int h = g % H_;

    const float* xb = x + (size_t)b * C_ * H_ + h;
    float xr[C_];
    float ss = 0.f;
    #pragma unroll
    for (int c = 0; c < C_; ++c) {
        float v = xb[(size_t)c * H_];   // coalesced across h
        xr[c] = v;
        ss += v * v;
    }
    const float nrm = sqrtf(ss);
    const float sc  = 1.f / fmaxf(nrm, 1e-12f);   // matches F.normalize eps

    _Float16* xo = xnh + (size_t)(b * H_ + h) * C_;
    #pragma unroll
    for (int c = 0; c < C_; ++c) {
        xr[c] *= sc;
        xo[c] = (_Float16)xr[c];
    }

    for (int m = 0; m < MID_; ++m) {
        float acc = g_b[m];
        #pragma unroll
        for (int c = 0; c < C_; ++c) acc += gw[m * C_ + c] * xr[c];
        vth[(size_t)(b * MID_ + m) * H_ + h] = (_Float16)acc;  // coalesced
    }
}

// -----------------------------------------------------------------------------
// Kernel 2: fused flash-style attention.
//   per wave: 16 query rows; loop key blocks of 32:
//     S(16x32) = Q(16x64) * K^T        -> 4x v_wmma_f32_16x16x32_f16
//     W = exp(exp(S))                  -> v_exp_f32 (range [1.44,15.2], safe:
//                                         no online-max rescaling needed)
//     acc(16x32) += W(16x32) * V(32x32)-> 2x v_wmma (W staged via wave-private
//                                         LDS to convert D-layout -> A-layout)
//     rowsum    += W(16x32) * ones     -> 1x v_wmma: rowsum lands in the same
//                                         D layout as acc -> elementwise divide
//   out2[q][m] = acc/rowsum
// -----------------------------------------------------------------------------
__global__ __launch_bounds__(256) void attn_kernel(
    const _Float16* __restrict__ xnh,  // [B,H,64]
    const _Float16* __restrict__ vth,  // [B,32,H]
    float* __restrict__ out2)          // [B,H,32]
{
    __shared__ _Float16 Wlds[8 * 16 * 32];   // 1 KB per wave, wave-private

    const int wave = threadIdx.x >> 5;
    const int lane = threadIdx.x & 31;
    const int nl   = lane & 15;        // column index within 16-wide tiles
    const int hh   = lane >> 4;        // lane half selects K sub-range
    const int b    = blockIdx.x >> 5;              // 32 query blocks per batch
    const int q0   = ((blockIdx.x & 31) << 7) + wave * 16;

    const _Float16* xb = xnh + (size_t)b * H_ * C_;
    const _Float16* vb = vth + (size_t)b * MID_ * H_;
    _Float16* wb = &Wlds[wave * (16 * 32)];

    // ---- Load query tile A once (16x64 f16, split into two K=32 chunks) ----
    // 16-bit A 16x32 layout (contiguous k, matching B): lane<16 holds row nl, K=0..15;
    // lane>=16 holds K=16..31  -> two 16B loads per chunk.
    v16h A0, A1;
    {
        const _Float16* qrow = xb + (size_t)(q0 + nl) * C_;
        v8h lo0 = *(const v8h*)(qrow +       hh * 16);
        v8h hi0 = *(const v8h*)(qrow + 8   + hh * 16);
        v8h lo1 = *(const v8h*)(qrow + 32  + hh * 16);
        v8h hi1 = *(const v8h*)(qrow + 40  + hh * 16);
        A0 = __builtin_shufflevector(lo0, hi0, 0,1,2,3,4,5,6,7,8,9,10,11,12,13,14,15);
        A1 = __builtin_shufflevector(lo1, hi1, 0,1,2,3,4,5,6,7,8,9,10,11,12,13,14,15);
    }

    // B operand of all-ones: rowsum[q] via the matrix pipe instead of shfl tree
    v16h ONES;
    #pragma unroll
    for (int i = 0; i < 16; ++i) ONES[i] = (_Float16)1.0f;

    v8f accA = {};                 // acc for m = 0..15   (C operand, persists)
    v8f accB = {};                 // acc for m = 16..31
    v8f rsD  = {};                 // rowsum (every column identical)

    for (int n0 = 0; n0 < H_; n0 += 32) {
        // ------------------ S = Q * K^T for two 16-key tiles ------------------
        v8f S[2];
        #pragma unroll
        for (int t = 0; t < 2; ++t) {
            // B 32x16 operand (contiguous k, matching A): lane holds column nl (= key row n0+16t+nl),
            // lane<16: K=0..15, lane>=16: K=16..31 -> 32B contiguous per lane.
            const _Float16* krow = xb + (size_t)(n0 + t * 16 + nl) * C_;
            v16h B0 = *(const v16h*)(krow +      hh * 16);   // c =  0..31
            v16h B1 = *(const v16h*)(krow + 32 + hh * 16);   // c = 32..63
            v8f s = {};
            s = __builtin_amdgcn_wmma_f32_16x16x32_f16(false, A0, false, B0,
                                                       (short)0, s, false, false);
            s = __builtin_amdgcn_wmma_f32_16x16x32_f16(false, A1, false, B1,
                                                       (short)0, s, false, false);
            S[t] = s;
        }

        // -------- W = exp(exp(S)); stage W to wave-private LDS ---------------
        // D layout: lane holds (q = r + 8*hh, n = nl) for r=0..7 per tile.
        #pragma unroll
        for (int r = 0; r < 8; ++r) {
            float w0 = __expf(__expf(S[0][r]));
            float w1 = __expf(__expf(S[1][r]));
            wb[(r + 8 * hh) * 32 + nl]      = (_Float16)w0;  // n = nl
            wb[(r + 8 * hh) * 32 + 16 + nl] = (_Float16)w1;  // n = 16+nl
        }

        // ---------------- reload W in f16 A layout from LDS ----------------
        v8h wlo = *(const v8h*)(wb + nl * 32 +     hh * 16);
        v8h whi = *(const v8h*)(wb + nl * 32 + 8 + hh * 16);
        v16h WA = __builtin_shufflevector(wlo, whi, 0,1,2,3,4,5,6,7,8,9,10,11,12,13,14,15);

        // ------------- V tiles as B operand (from transposed V) -------------
        const _Float16* v0p = vb + (size_t)(nl)      * H_ + n0 + hh * 16;
        const _Float16* v1p = vb + (size_t)(16 + nl) * H_ + n0 + hh * 16;
        v16h BV0 = *(const v16h*)v0p;
        v16h BV1 = *(const v16h*)v1p;

        accA = __builtin_amdgcn_wmma_f32_16x16x32_f16(false, WA, false, BV0,
                                                      (short)0, accA, false, false);
        accB = __builtin_amdgcn_wmma_f32_16x16x32_f16(false, WA, false, BV1,
                                                      (short)0, accB, false, false);
        rsD  = __builtin_amdgcn_wmma_f32_16x16x32_f16(false, WA, false, ONES,
                                                      (short)0, rsD,  false, false);
    }

    // -------- epilogue: out2[q][m] = acc/rowsum (layouts already match) -----
    #pragma unroll
    for (int r = 0; r < 8; ++r) {
        const float inv = 1.f / rsD[r];    // rowsum >= 32*exp(1/e) > 0 always
        const int q = q0 + r + 8 * hh;
        float* o = out2 + ((size_t)b * H_ + q) * MID_;
        o[nl]      = accA[r] * inv;
        o[16 + nl] = accB[r] * inv;
    }
}

// -----------------------------------------------------------------------------
// Kernel 3: output projection + residual.  out = patch + o_w*out2 + o_b
// -----------------------------------------------------------------------------
__global__ __launch_bounds__(256) void out_kernel(
    const float* __restrict__ patch,  // [B,C,H]
    const float* __restrict__ o_w,    // [C,MID]
    const float* __restrict__ o_b,    // [C]
    const float* __restrict__ out2,   // [B,H,32]
    float* __restrict__ out)          // [B,C,H]
{
    __shared__ float ow[C_ * MID_];   // 8 KB
    for (int i = threadIdx.x; i < C_ * MID_; i += 256) ow[i] = o_w[i];
    __syncthreads();

    const int g = blockIdx.x * 256 + threadIdx.x;
    const int b = g / H_;
    const int h = g % H_;

    float t[MID_];
    const float* trow = out2 + (size_t)(b * H_ + h) * MID_;
    #pragma unroll
    for (int m = 0; m < MID_; ++m) t[m] = trow[m];

    const float* pb = patch + (size_t)b * C_ * H_ + h;
    float*       ob = out   + (size_t)b * C_ * H_ + h;
    for (int c = 0; c < C_; ++c) {
        float acc = o_b[c];
        #pragma unroll
        for (int m = 0; m < MID_; ++m) acc += ow[c * MID_ + m] * t[m];
        ob[(size_t)c * H_] = pb[(size_t)c * H_] + acc;   // coalesced across h
    }
}

// -----------------------------------------------------------------------------
extern "C" void kernel_launch(void* const* d_in, const int* in_sizes, int n_in,
                              void* d_out, int out_size, void* d_ws, size_t ws_size,
                              hipStream_t stream) {
    (void)in_sizes; (void)n_in; (void)out_size; (void)ws_size;

    const float* patch = (const float*)d_in[0];   // [B,C,H,1]
    const float* g_w   = (const float*)d_in[1];   // [MID,C]
    const float* g_b   = (const float*)d_in[2];   // [MID]
    const float* o_w   = (const float*)d_in[3];   // [C,MID]
    const float* o_b   = (const float*)d_in[4];   // [C]

    // Workspace: xnh (4 MB f16) | vth (2 MB f16) | out2 (4 MB f32)  = 10 MB
    _Float16* xnh  = (_Float16*)d_ws;
    _Float16* vth  = xnh + (size_t)B_ * H_ * C_;
    float*    out2 = (float*)(vth + (size_t)B_ * MID_ * H_);

    prep_kernel<<<B_ * H_ / 256, 256, 0, stream>>>(patch, g_w, g_b, xnh, vth);
    attn_kernel<<<B_ * (H_ / 128), 256, 0, stream>>>(xnh, vth, out2);
    out_kernel<<<B_ * H_ / 256, 256, 0, stream>>>(patch, o_w, o_b, out2,
                                                  (float*)d_out);
}
